// LSTMModel_20444044329612
// MI455X (gfx1250) — matrix operand order repaired and checked by reading, not compiled, not run on hardware
//
#include <hip/hip_runtime.h>

// ---------------------------------------------------------------------------
// LSTM (B=128, T=1024, D=128, H=256, O=64) for MI455X / gfx1250.
//
// Round 3: split grid barrier (signal / wait). Per step the critical path is
// only the h@U half (8 chunks); the x@W half for step t+1 is computed after
// signaling, hidden behind the barrier. Fragment loads (LDS B, global A) are
// software-pipelined one chunk ahead so WMMAs don't drain dscnt to 0.
// Weights live in LDS (48 KB/block); cell state in VGPRs for all 1024 steps.
// ---------------------------------------------------------------------------

typedef __attribute__((ext_vector_type(16))) __bf16 v16bf;
typedef __attribute__((ext_vector_type(8)))  float  v8f;

#define WMMA_BF16(a, b, c) \
  __builtin_amdgcn_wmma_f32_16x16x32_bf16(false, (a), false, (b), (short)0, (c), false, false)

#define BB 128
#define TT 1024
#define DD 128
#define HH 256
#define OO 64

// ---- workspace layout (bytes), all offsets 256-aligned -------------------
#define OFF_SYNC   ((size_t)0)                      // 2 x u32 barrier state
#define OFF_WU     ((size_t)256)                    // [64 ntile][12 kchunk][32 lane][16e] bf16 = 786432
#define OFF_WOUT   (OFF_WU + (size_t)786432)        // [4][8][32][16] bf16 = 32768
#define OFF_HBUF   (OFF_WOUT + (size_t)32768)       // 2 x [8 mtile][8 kchunk][32][16] bf16 = 131072
#define OFF_STEPX  (OFF_HBUF + (size_t)131072)      // [8][4][32][16] bf16 = 32768
#define OFF_STEPH  (OFF_STEPX + (size_t)32768)      // [8][8][32][16] bf16 = 65536
#define OFF_XPACK  (OFF_STEPH + (size_t)65536)      // [1024 t][8][4][32][16] bf16 = 33554432
#define OFF_HS     (OFF_XPACK + (size_t)33554432)   // [1025 slot][8][8][32][16] bf16 = 67174400

// ---- d_out layout (floats) ----------------------------------------------
#define OUT_ALL   ((size_t)0)
#define OUT_STEP  ((size_t)BB * TT * OO)            // 8388608
#define OUT_SH    (OUT_STEP + (size_t)BB * OO)      // 8396800
#define OUT_SC    (OUT_SH + (size_t)BB * HH)        // 8429568
#define OUT_H0    (OUT_SC + (size_t)BB * HH)        // 8462336
#define OUT_C0    (OUT_H0 + (size_t)BB * HH)        // 8495104

__device__ __forceinline__ unsigned short f2bf(float f) {
  unsigned int u = __float_as_uint(f);
  u += 0x7FFFu + ((u >> 16) & 1u);                  // round-to-nearest-even
  return (unsigned short)(u >> 16);
}

// A-fragment (16-bit, 16x32) k index for element e, lane-half h (ISA 7.12.2)
__device__ __forceinline__ int ka_of(int e, int half) {
  return e + 8 * half + (e >= 8 ? 8 : 0);
}

// ---- split grid barrier --------------------------------------------------
__device__ __forceinline__ void grid_signal(unsigned int* counter, unsigned int* generation,
                                            unsigned int nblk) {
  __syncthreads();                                   // block done writing h(t+1)
  if (threadIdx.x == 0) {
    unsigned int g = __hip_atomic_load(generation, __ATOMIC_RELAXED, __HIP_MEMORY_SCOPE_AGENT);
    unsigned int prev = __hip_atomic_fetch_add(counter, 1u, __ATOMIC_ACQ_REL, __HIP_MEMORY_SCOPE_AGENT);
    if (prev == nblk - 1u) {
      __hip_atomic_store(counter, 0u, __ATOMIC_RELAXED, __HIP_MEMORY_SCOPE_AGENT);
      __hip_atomic_store(generation, g + 1u, __ATOMIC_RELEASE, __HIP_MEMORY_SCOPE_AGENT);
    }
  }
}

__device__ __forceinline__ void grid_wait(unsigned int* generation, unsigned int target) {
  if (threadIdx.x == 0) {
    while (__hip_atomic_load(generation, __ATOMIC_ACQUIRE, __HIP_MEMORY_SCOPE_AGENT) < target) {
      __builtin_amdgcn_s_sleep(1);
    }
  }
  __syncthreads();
}

// ---------------------------------------------------------------------------
// K1: pack weights/inputs to bf16 WMMA-fragment layouts, init barrier, h0/c0
// ---------------------------------------------------------------------------
__global__ void pack_init_kernel(const float* __restrict__ all_input,
                                 const float* __restrict__ step_input,
                                 const float* __restrict__ step_hidden,
                                 const float* __restrict__ W,
                                 const float* __restrict__ U,
                                 const float* __restrict__ Wout,
                                 const float* __restrict__ init_h,
                                 const float* __restrict__ init_c,
                                 float* __restrict__ out,
                                 char* __restrict__ ws) {
  const int tid = blockIdx.x * blockDim.x + threadIdx.x;
  const int nthr = gridDim.x * blockDim.x;
  if (tid == 0) {
    unsigned int* s = (unsigned int*)(ws + OFF_SYNC);
    s[0] = 0u; s[1] = 0u;
  }
  // W||U combined B-matrix, K = 0..127 -> W rows, K = 128..383 -> U rows
  unsigned short* wu = (unsigned short*)(ws + OFF_WU);
  for (int i = tid; i < 64 * 12 * 512; i += nthr) {
    int e = i & 15, lane = (i >> 4) & 31, chunk = (i >> 9) % 12, ntile = i / 6144;
    int half = lane >> 4, n = lane & 15;
    int K = chunk * 32 + ka_of(e, half);          // B-fragment k mapping
    int col = ntile * 16 + n;
    float v = (K < DD) ? W[K * (4 * HH) + col] : U[(K - DD) * (4 * HH) + col];
    wu[i] = f2bf(v);
  }
  unsigned short* wo = (unsigned short*)(ws + OFF_WOUT);
  for (int i = tid; i < 4 * 8 * 512; i += nthr) {
    int e = i & 15, lane = (i >> 4) & 31, chunk = (i >> 9) & 7, ntile = i >> 12;
    int half = lane >> 4, n = lane & 15;
    int K = chunk * 32 + ka_of(e, half);
    wo[i] = f2bf(Wout[K * OO + ntile * 16 + n]);
  }
  unsigned short* xp = (unsigned short*)(ws + OFF_XPACK);
  for (long i = tid; i < (long)TT * 8 * 4 * 512; i += nthr) {
    int e = (int)(i & 15), lane = (int)((i >> 4) & 31);
    int chunk = (int)((i >> 9) & 3), mtile = (int)((i >> 11) & 7);
    int t = (int)(i >> 14);
    int half = lane >> 4, ml = lane & 15;
    int k = chunk * 32 + ka_of(e, half);
    int b = mtile * 16 + ml;
    xp[i] = f2bf(all_input[((long)b * TT + t) * DD + k]);
  }
  unsigned short* sx = (unsigned short*)(ws + OFF_STEPX);
  for (int i = tid; i < 8 * 4 * 512; i += nthr) {
    int e = i & 15, lane = (i >> 4) & 31, chunk = (i >> 9) & 3, mtile = i >> 11;
    int half = lane >> 4, ml = lane & 15;
    int k = chunk * 32 + ka_of(e, half);
    sx[i] = f2bf(step_input[(mtile * 16 + ml) * DD + k]);
  }
  unsigned short* shp = (unsigned short*)(ws + OFF_STEPH);
  for (int i = tid; i < 8 * 8 * 512; i += nthr) {
    int e = i & 15, lane = (i >> 4) & 31, chunk = (i >> 9) & 7, mtile = i >> 12;
    int half = lane >> 4, ml = lane & 15;
    int k = chunk * 32 + ka_of(e, half);
    shp[i] = f2bf(step_hidden[(mtile * 16 + ml) * HH + k]);
  }
  unsigned short* hb0 = (unsigned short*)(ws + OFF_HBUF);   // parity-0 h buffer
  for (int i = tid; i < 8 * 8 * 512; i += nthr) {
    int e = i & 15, lane = (i >> 4) & 31, chunk = (i >> 9) & 7;
    int half = lane >> 4;
    int k = chunk * 32 + ka_of(e, half);
    hb0[i] = f2bf(init_h[k]);                     // broadcast over batch
  }
  for (int i = tid; i < BB * HH; i += nthr) {
    out[OUT_H0 + i] = init_h[i % HH];
    out[OUT_C0 + i] = init_c[i % HH];
  }
}

// ---------------------------------------------------------------------------
// K2: persistent recurrent scan. 16 blocks x 256 threads = 128 waves.
// Block == h_tile; the block's 48 weight B-fragments live in LDS.
// Iteration: wait(h(t)) -> h-part WMMAs -> gates -> write h(t+1) -> signal
//            -> x-part WMMAs for t+1 (hidden behind barrier).
// ---------------------------------------------------------------------------
__global__ __launch_bounds__(256) void lstm_scan_kernel(const float* __restrict__ bias,
                                                        const float* __restrict__ init_c,
                                                        char* __restrict__ ws) {
  const int lane = threadIdx.x & 31;
  const int m_tile = threadIdx.x >> 5;                   // 0..7 (wave in block)
  const int h_tile = blockIdx.x;                         // 0..15 (block)
  const int half = lane >> 4;
  const int n = lane & 15;
  const int col = h_tile * 16 + n;

  const v16bf* __restrict__ WU = (const v16bf*)(ws + OFF_WU);
  const v16bf* __restrict__ XP = (const v16bf*)(ws + OFF_XPACK);
  unsigned short* __restrict__ HS = (unsigned short*)(ws + OFF_HS);
  unsigned int* cnt = (unsigned int*)(ws + OFF_SYNC);
  unsigned int* gen = cnt + 1;

  // ---- stage this block's 48 B-fragments (4 gates x 12 chunks) in LDS ----
  __shared__ v16bf smB[4 * 12 * 32];                     // 48 KB
#pragma unroll
  for (int it = 0; it < 6; ++it) {
    int i = it * 256 + threadIdx.x;                      // 0..1535
    int g = i / 384, rem = i % 384, c = rem >> 5, l = rem & 31;
    smB[i] = WU[((g * 16 + h_tile) * 12 + c) * 32 + l];
  }
  __syncthreads();

  const float b_i = bias[col];
  const float b_f = bias[HH + col];
  const float b_g = bias[2 * HH + col];
  const float b_o = bias[3 * HH + col];

  v8f cc;
  {
    float c0 = init_c[col];
#pragma unroll
    for (int r = 0; r < 8; ++r) cc[r] = c0;
  }

  // fragment destination mapping for writing h (A-layout inverse)
  const int chunk_h = h_tile >> 1;
  const int k32 = ((h_tile & 1) << 4) | n;
  int half_a, e_dst;
  if (k32 < 8)       { half_a = 0; e_dst = k32; }
  else if (k32 < 16) { half_a = 1; e_dst = k32 - 8; }
  else if (k32 < 24) { half_a = 0; e_dst = k32 - 8; }
  else               { half_a = 1; e_dst = k32 - 16; }

  v8f ai, af, ag, ao;

  // x-part for step t: assumes ax0..ax3 hold step t's x A-fragments; resets
  // and accumulates ai..ao with chunks 0..3, B loads pipelined one chunk.
#define X_PART(AX0, AX1, AX2, AX3)                                          \
  do {                                                                      \
    v16bf b0 = smB[(0 * 12 + 0) * 32 + lane];                               \
    v16bf b1 = smB[(1 * 12 + 0) * 32 + lane];                               \
    v16bf b2 = smB[(2 * 12 + 0) * 32 + lane];                               \
    v16bf b3 = smB[(3 * 12 + 0) * 32 + lane];                               \
    ai = {}; af = {}; ag = {}; ao = {};                                     \
    _Pragma("unroll")                                                       \
    for (int c = 0; c < 4; ++c) {                                           \
      v16bf a = (c == 0) ? AX0 : (c == 1) ? AX1 : (c == 2) ? AX2 : AX3;     \
      v16bf n0 = b0, n1 = b1, n2 = b2, n3 = b3;                             \
      if (c < 3) {                                                          \
        n0 = smB[(0 * 12 + c + 1) * 32 + lane];                             \
        n1 = smB[(1 * 12 + c + 1) * 32 + lane];                             \
        n2 = smB[(2 * 12 + c + 1) * 32 + lane];                             \
        n3 = smB[(3 * 12 + c + 1) * 32 + lane];                             \
      }                                                                     \
      ai = WMMA_BF16(a, b0, ai);                                            \
      af = WMMA_BF16(a, b1, af);                                            \
      ag = WMMA_BF16(a, b2, ag);                                            \
      ao = WMMA_BF16(a, b3, ao);                                            \
      b0 = n0; b1 = n1; b2 = n2; b3 = n3;                                   \
    }                                                                       \
  } while (0)

  // prologue: x-part for t = 0
  v16bf ax0 = XP[((long)m_tile * 4 + 0) * 32 + lane];
  v16bf ax1 = XP[((long)m_tile * 4 + 1) * 32 + lane];
  v16bf ax2 = XP[((long)m_tile * 4 + 2) * 32 + lane];
  v16bf ax3 = XP[((long)m_tile * 4 + 3) * 32 + lane];
  X_PART(ax0, ax1, ax2, ax3);

  for (int t = 0; t < TT; ++t) {
    const v16bf* hsrc = (const v16bf*)(ws + OFF_HBUF + (size_t)(t & 1) * 65536);
    unsigned short* hdst = (unsigned short*)(ws + OFF_HBUF + (size_t)((t + 1) & 1) * 65536);

    grid_wait(gen, (unsigned int)t);                 // h(t) visible

    // h @ U part (chunks 4..11), A from global double-buffer, B from LDS,
    // both pipelined one chunk ahead.
    {
      v16bf a  = hsrc[(m_tile * 8 + 0) * 32 + lane];
      v16bf b0 = smB[(0 * 12 + 4) * 32 + lane];
      v16bf b1 = smB[(1 * 12 + 4) * 32 + lane];
      v16bf b2 = smB[(2 * 12 + 4) * 32 + lane];
      v16bf b3 = smB[(3 * 12 + 4) * 32 + lane];
#pragma unroll
      for (int c = 0; c < 8; ++c) {
        v16bf an = a, n0 = b0, n1 = b1, n2 = b2, n3 = b3;
        if (c < 7) {
          an = hsrc[(m_tile * 8 + c + 1) * 32 + lane];
          n0 = smB[(0 * 12 + 5 + c) * 32 + lane];
          n1 = smB[(1 * 12 + 5 + c) * 32 + lane];
          n2 = smB[(2 * 12 + 5 + c) * 32 + lane];
          n3 = smB[(3 * 12 + 5 + c) * 32 + lane];
        }
        ai = WMMA_BF16(a, b0, ai);
        af = WMMA_BF16(a, b1, af);
        ag = WMMA_BF16(a, b2, ag);
        ao = WMMA_BF16(a, b3, ao);
        a = an; b0 = n0; b1 = n1; b2 = n2; b3 = n3;
      }
    }

#pragma unroll
    for (int r = 0; r < 8; ++r) {
      float iv = 1.0f / (1.0f + __expf(-(ai[r] + b_i)));
      float fv = 1.0f / (1.0f + __expf(-(af[r] + b_f)));
      float gv = tanhf(ag[r] + b_g);
      float ov = 1.0f / (1.0f + __expf(-(ao[r] + b_o)));
      float cv = fv * cc[r] + iv * gv;
      cc[r] = cv;
      float hv = ov * tanhf(cv);
      unsigned short hb = f2bf(hv);
      int lane_a = (r + 8 * half) + 16 * half_a;
      int fidx = ((m_tile * 8 + chunk_h) * 32 + lane_a) * 16 + e_dst;
      hdst[fidx] = hb;                         // next step's A fragments
      HS[(long)t * 32768 + fidx] = hb;         // hs for projection
    }

    grid_signal(cnt, gen, gridDim.x);          // announce h(t+1)

    // x-part for step t+1: runs in the barrier's shadow
    if (t + 1 < TT) {
      ax0 = XP[(((long)(t + 1) * 8 + m_tile) * 4 + 0) * 32 + lane];
      ax1 = XP[(((long)(t + 1) * 8 + m_tile) * 4 + 1) * 32 + lane];
      ax2 = XP[(((long)(t + 1) * 8 + m_tile) * 4 + 2) * 32 + lane];
      ax3 = XP[(((long)(t + 1) * 8 + m_tile) * 4 + 3) * 32 + lane];
      X_PART(ax0, ax1, ax2, ax3);
    }
  }
#undef X_PART
}

// ---------------------------------------------------------------------------
// K4: single LSTM step (sh, sc) -> d_out + HS slot 1024 for projection
// ---------------------------------------------------------------------------
__global__ __launch_bounds__(256) void lstm_step_kernel(const float* __restrict__ bias,
                                                        const float* __restrict__ step_cell,
                                                        float* __restrict__ out,
                                                        char* __restrict__ ws) {
  const int lane = threadIdx.x & 31;
  const int m_tile = threadIdx.x >> 5;
  const int h_tile = blockIdx.x;
  const int half = lane >> 4;
  const int n = lane & 15;
  const int col = h_tile * 16 + n;

  const v16bf* __restrict__ WU = (const v16bf*)(ws + OFF_WU);
  const v16bf* __restrict__ SX = (const v16bf*)(ws + OFF_STEPX);
  const v16bf* __restrict__ SH = (const v16bf*)(ws + OFF_STEPH);
  unsigned short* __restrict__ HS = (unsigned short*)(ws + OFF_HS);

  const float b_i = bias[col];
  const float b_f = bias[HH + col];
  const float b_g = bias[2 * HH + col];
  const float b_o = bias[3 * HH + col];

  v8f ai = {}; v8f af = {}; v8f ag = {}; v8f ao = {};
#pragma unroll
  for (int c = 0; c < 12; ++c) {
    v16bf a = (c < 4) ? SX[(m_tile * 4 + c) * 32 + lane]
                      : SH[(m_tile * 8 + (c - 4)) * 32 + lane];
    ai = WMMA_BF16(a, WU[(( 0 + h_tile) * 12 + c) * 32 + lane], ai);
    af = WMMA_BF16(a, WU[((16 + h_tile) * 12 + c) * 32 + lane], af);
    ag = WMMA_BF16(a, WU[((32 + h_tile) * 12 + c) * 32 + lane], ag);
    ao = WMMA_BF16(a, WU[((48 + h_tile) * 12 + c) * 32 + lane], ao);
  }

  const int chunk_h = h_tile >> 1;
  const int k32 = ((h_tile & 1) << 4) | n;
  int half_a, e_dst;
  if (k32 < 8)       { half_a = 0; e_dst = k32; }
  else if (k32 < 16) { half_a = 1; e_dst = k32 - 8; }
  else if (k32 < 24) { half_a = 0; e_dst = k32 - 8; }
  else               { half_a = 1; e_dst = k32 - 16; }

#pragma unroll
  for (int r = 0; r < 8; ++r) {
    int brow = m_tile * 16 + r + 8 * half;
    float cold = step_cell[brow * HH + col];
    float iv = 1.0f / (1.0f + __expf(-(ai[r] + b_i)));
    float fv = 1.0f / (1.0f + __expf(-(af[r] + b_f)));
    float gv = tanhf(ag[r] + b_g);
    float ov = 1.0f / (1.0f + __expf(-(ao[r] + b_o)));
    float cv = fv * cold + iv * gv;
    float hv = ov * tanhf(cv);
    out[OUT_SC + (size_t)brow * HH + col] = cv;
    out[OUT_SH + (size_t)brow * HH + col] = hv;
    int lane_a = (r + 8 * half) + 16 * half_a;
    int fidx = ((m_tile * 8 + chunk_h) * 32 + lane_a) * 16 + e_dst;
    HS[(long)TT * 32768 + fidx] = f2bf(hv);    // slot 1024 = sh
  }
}

// ---------------------------------------------------------------------------
// K3: projection  hs @ W_out + b_out  (slots 0..1023 -> all_output,
//     slot 1024 -> step_output). W_out fragments staged in LDS per block.
// ---------------------------------------------------------------------------
__global__ __launch_bounds__(256) void proj_kernel(const float* __restrict__ b_out,
                                                   float* __restrict__ out,
                                                   char* __restrict__ ws) {
  const int lane = threadIdx.x & 31;
  const int wid = blockIdx.x * 8 + (threadIdx.x >> 5);
  const int slot = wid >> 3;
  const int m_tile = wid & 7;
  const int half = lane >> 4;
  const int n = lane & 15;

  const v16bf* __restrict__ HS = (const v16bf*)(ws + OFF_HS);
  const v16bf* __restrict__ WO = (const v16bf*)(ws + OFF_WOUT);

  __shared__ v16bf smWO[4 * 8 * 32];                     // 32 KB, block-shared
#pragma unroll
  for (int it = 0; it < 4; ++it) {
    int i = it * 256 + threadIdx.x;                      // 0..1023
    smWO[i] = WO[i];
  }
  __syncthreads();
  if (slot >= TT + 1) return;

  v8f a0 = {}; v8f a1 = {}; v8f a2 = {}; v8f a3 = {};
#pragma unroll
  for (int c = 0; c < 8; ++c) {
    v16bf a = HS[((long)slot * 64 + m_tile * 8 + c) * 32 + lane];
    a0 = WMMA_BF16(a, smWO[(0 * 8 + c) * 32 + lane], a0);
    a1 = WMMA_BF16(a, smWO[(1 * 8 + c) * 32 + lane], a1);
    a2 = WMMA_BF16(a, smWO[(2 * 8 + c) * 32 + lane], a2);
    a3 = WMMA_BF16(a, smWO[(3 * 8 + c) * 32 + lane], a3);
  }
  const float bo0 = b_out[n], bo1 = b_out[16 + n], bo2 = b_out[32 + n], bo3 = b_out[48 + n];
#pragma unroll
  for (int r = 0; r < 8; ++r) {
    int brow = m_tile * 16 + r + 8 * half;
    if (slot < TT) {
      size_t base = ((size_t)brow * TT + slot) * OO;
      out[base + n]      = a0[r] + bo0;
      out[base + 16 + n] = a1[r] + bo1;
      out[base + 32 + n] = a2[r] + bo2;
      out[base + 48 + n] = a3[r] + bo3;
    } else {
      size_t base = OUT_STEP + (size_t)brow * OO;
      out[base + n]      = a0[r] + bo0;
      out[base + 16 + n] = a1[r] + bo1;
      out[base + 32 + n] = a2[r] + bo2;
      out[base + 48 + n] = a3[r] + bo3;
    }
  }
}

// ---------------------------------------------------------------------------
extern "C" void kernel_launch(void* const* d_in, const int* in_sizes, int n_in,
                              void* d_out, int out_size, void* d_ws, size_t ws_size,
                              hipStream_t stream) {
  (void)in_sizes; (void)n_in; (void)out_size; (void)ws_size;
  const float* all_input   = (const float*)d_in[0];
  const float* step_input  = (const float*)d_in[1];
  const float* step_hidden = (const float*)d_in[2];
  const float* step_cell   = (const float*)d_in[3];
  const float* W      = (const float*)d_in[4];
  const float* U      = (const float*)d_in[5];
  const float* bias   = (const float*)d_in[6];
  const float* W_out  = (const float*)d_in[7];
  const float* b_out  = (const float*)d_in[8];
  const float* init_h = (const float*)d_in[9];
  const float* init_c = (const float*)d_in[10];
  float* out = (float*)d_out;
  char* ws = (char*)d_ws;

  pack_init_kernel<<<2048, 256, 0, stream>>>(all_input, step_input, step_hidden,
                                             W, U, W_out, init_h, init_c, out, ws);
  lstm_step_kernel<<<16, 256, 0, stream>>>(bias, step_cell, out, ws);
  lstm_scan_kernel<<<16, 256, 0, stream>>>(bias, init_c, ws);
  proj_kernel<<<(TT + 1 + 7) / 8, 256, 0, stream>>>(b_out, out, ws);
}
